// DAGNet_85813446574514
// MI455X (gfx1250) — matrix operand order repaired and checked by reading, not compiled, not run on hardware
//
#include <hip/hip_runtime.h>
#include <hip/hip_bf16.h>
#include <math.h>

typedef _Float16 h16;
typedef __attribute__((ext_vector_type(16))) _Float16 v16h;
typedef __attribute__((ext_vector_type(8)))  _Float16 v8h;
typedef __attribute__((ext_vector_type(8)))  float    v8f;

#define T_STEPS 30
#define BATCH   2048
#define SBLK    128
#define SLOPE   0.01f

// ---------------------------------------------------------------- utilities

__device__ __forceinline__ float wave_sum(float v) {
#pragma unroll
  for (int i = 16; i >= 1; i >>= 1) v += __shfl_xor(v, i, 32);
  return v;
}
__device__ __forceinline__ float wave_max(float v) {
#pragma unroll
  for (int i = 16; i >= 1; i >>= 1) v = fmaxf(v, __shfl_xor(v, i, 32));
  return v;
}

__device__ __forceinline__ float apply_act(float v, int act) {
  if (act == 1)      return v > 0.0f ? v : 0.0f;       // relu
  else if (act == 2) return v > 0.0f ? v : SLOPE * v;  // leaky relu
  return v;
}

// ------------------------------------------------------ WMMA GEMM  C = act(A@W + b)
// A: [BATCH x Kpad] f16 row-major (Kpad % 32 == 0, zero padded)
// W: [Kpad x N]     f16 row-major (N % 64 == 0)
// grid = (N/64, BATCH/16), block = 32: one wave computes a 16x64 tile
// (A fragment loaded once per K-step and reused across 4 WMMAs -> 4x less A traffic)
__global__ void gemm_wmma_kernel(const h16* __restrict__ A, const h16* __restrict__ W,
                                 const float* __restrict__ bias,
                                 float* __restrict__ C, h16* __restrict__ Ch,
                                 int Kpad, int N, int act) {
  const int lane  = threadIdx.x;
  const int nt4   = blockIdx.x;               // 64-wide N tile
  const int mtile = blockIdx.y;
  const int mrow  = lane & 15;
  const int koff  = (lane >> 4) * 8;          // A-fragment K sub-offset per ISA layout
  const h16* arow = A + (size_t)(mtile * 16 + mrow) * Kpad;
  const h16* wcol = W + nt4 * 64;

  v8f acc0 = {}, acc1 = {}, acc2 = {}, acc3 = {};
  for (int kb = 0; kb < Kpad; kb += 32) {
    // A fragment: VGPRs 0-3 = K[koff..koff+7], VGPRs 4-7 = K[16+koff..16+koff+7]
    v8h alo = *(const v8h*)(arow + kb + koff);
    v8h ahi = *(const v8h*)(arow + kb + 16 + koff);
    v16h a;
#pragma unroll
    for (int i = 0; i < 8; ++i) { a[i] = alo[i]; a[i + 8] = ahi[i]; }
    // B fragments: lane = N column, elements = K in the same order as A, 4 x 16 N columns
    v16h b0, b1, b2, b3;
#pragma unroll
    for (int i = 0; i < 8; ++i) {
      const h16* p0 = wcol + (size_t)(kb + koff + i) * N + mrow;
      const h16* p1 = wcol + (size_t)(kb + 16 + koff + i) * N + mrow;
      b0[i] = p0[0]; b1[i] = p0[16]; b2[i] = p0[32]; b3[i] = p0[48];
      b0[i + 8] = p1[0]; b1[i + 8] = p1[16]; b2[i + 8] = p1[32]; b3[i + 8] = p1[48];
    }
    acc0 = __builtin_amdgcn_wmma_f32_16x16x32_f16(false, a, false, b0, (short)0, acc0, false, false);
    acc1 = __builtin_amdgcn_wmma_f32_16x16x32_f16(false, a, false, b1, (short)0, acc1, false, false);
    acc2 = __builtin_amdgcn_wmma_f32_16x16x32_f16(false, a, false, b2, (short)0, acc2, false, false);
    acc3 = __builtin_amdgcn_wmma_f32_16x16x32_f16(false, a, false, b3, (short)0, acc3, false, false);
  }
  const int mmbase = mtile * 16 + 8 * (lane >> 4);
  const int nlane  = lane & 15;
#define GEMM_EPI(ACC, J)                                                      \
  {                                                                           \
    const int nn = nt4 * 64 + (J) * 16 + nlane;                               \
    const float bv = bias ? bias[nn] : 0.0f;                                  \
    _Pragma("unroll")                                                         \
    for (int r = 0; r < 8; ++r) {                                             \
      int mm = mmbase + r;                                                    \
      float v = apply_act(ACC[r] + bv, act);                                  \
      C[(size_t)mm * N + nn] = v;                                             \
      if (Ch) Ch[(size_t)mm * N + nn] = (h16)v;                               \
    }                                                                         \
  }
  GEMM_EPI(acc0, 0) GEMM_EPI(acc1, 1) GEMM_EPI(acc2, 2) GEMM_EPI(acc3, 3)
#undef GEMM_EPI
}

// ----------------------------------------- block-diagonal GCN multiply (adj @ X)
// adj_t: [BATCH x BATCH] f32, only 16x16 diagonal blocks non-zero
// X    : [BATCH x N] f32;   Y = maybe_relu(adj@X + bias)
// grid = (N/64, SBLK), block = 32; adjacency fragment converted once, reused 4x
__global__ void gcn_wmma_kernel(const float* __restrict__ adj, const float* __restrict__ X,
                                const float* __restrict__ bias,
                                float* __restrict__ Y, h16* __restrict__ Yh,
                                int N, int do_relu) {
  const int lane = threadIdx.x;
  const int nt4  = blockIdx.x;
  const int s    = blockIdx.y;
  const int mrow = lane & 15;
  const int koff = (lane >> 4) * 8;
  const float* arow = adj + (size_t)(s * 16 + mrow) * BATCH + s * 16;
  v16h a;
#pragma unroll
  for (int i = 0; i < 8; ++i) { a[i] = (h16)arow[koff + i]; a[i + 8] = (h16)0.0f; }

  const float* xrow = X + (size_t)(s * 16 + koff) * N + nt4 * 64 + (lane & 15);
  const int mmbase = s * 16 + 8 * (lane >> 4);
  const int nlane  = lane & 15;
#pragma unroll
  for (int j = 0; j < 4; ++j) {
    v16h b;
#pragma unroll
    for (int e = 0; e < 8; ++e) { b[e] = (h16)xrow[(size_t)e * N + j * 16]; b[e + 8] = (h16)0.0f; }
    v8f acc = {};
    acc = __builtin_amdgcn_wmma_f32_16x16x32_f16(false, a, false, b,
                                                 (short)0, acc, false, false);
    const int nn = nt4 * 64 + j * 16 + nlane;
    const float bv = bias ? bias[nn] : 0.0f;
#pragma unroll
    for (int r = 0; r < 8; ++r) {
      int mm = mmbase + r;
      float v = acc[r] + bv;
      if (do_relu) v = v > 0.0f ? v : 0.0f;
      Y[(size_t)mm * N + nn] = v;
      if (Yh) Yh[(size_t)mm * N + nn] = (h16)v;
    }
  }
}

// -------------------------------------------------------------- small kernels

// concat up to 4 f32 sources into zero-padded f16 activation matrix [BATCH x Kpad]
__global__ void pack_cat_kernel(const float* s0, int w0, const float* s1, int w1,
                                const float* s2, int w2, const float* s3, int w3,
                                h16* A, int Kpad) {
  int idx = blockIdx.x * blockDim.x + threadIdx.x;
  if (idx >= BATCH * Kpad) return;
  int row = idx / Kpad, k = idx % Kpad;
  float v = 0.0f;
  if (k < w0) v = s0[row * w0 + k];
  else { k -= w0;
    if (k < w1) v = s1[row * w1 + k];
    else { k -= w1;
      if (k < w2) v = s2[row * w2 + k];
      else { k -= w2;
        if (k < w3) v = s3[row * w3 + k];
      } } }
  A[idx] = (h16)v;
}

// weight f32 [K x N] -> f16 [Kpad x N] (zero pad extra K rows)
__global__ void wconv_kernel(const float* __restrict__ W, h16* __restrict__ Wh,
                             int K, int N, int Kpad) {
  int idx = blockIdx.x * blockDim.x + threadIdx.x;
  if (idx >= Kpad * N) return;
  int k = idx / N;
  Wh[idx] = (k < K) ? (h16)W[idx] : (h16)0.0f;
}

// d_t[b][j] = traj_t[(b/16)*16 + j/2][j%2]  (32 contiguous floats per 16-agent group)
__global__ void dt_kernel(const float* __restrict__ traj_t, float* __restrict__ dt) {
  int idx = blockIdx.x * blockDim.x + threadIdx.x;
  if (idx >= BATCH * 32) return;
  int b = idx >> 5, j = idx & 31;
  dt[idx] = traj_t[(size_t)(b >> 4) * 32 + j];
}

// row softmax over 64 cols; one wave per row; optional f16 mirror
__global__ void softmax64_kernel(const float* __restrict__ X, float* __restrict__ Y,
                                 h16* __restrict__ Yh) {
  int row = blockIdx.x, lane = threadIdx.x;
  float v0 = X[row * 64 + lane], v1 = X[row * 64 + 32 + lane];
  float m = wave_max(fmaxf(v0, v1));
  float e0 = expf(v0 - m), e1 = expf(v1 - m);
  float s = wave_sum(e0 + e1);
  float r0 = e0 / s, r1 = e1 / s;
  Y[row * 64 + lane] = r0; Y[row * 64 + 32 + lane] = r1;
  if (Yh) { Yh[row * 64 + lane] = (h16)r0; Yh[row * 64 + 32 + lane] = (h16)r1; }
}

// z = eps*exp(0.5*logvar)+mean, written straight to f16 A-matrix
__global__ void zpack_kernel(const float* __restrict__ eps_t, const float* __restrict__ el,
                             const float* __restrict__ em, h16* __restrict__ zh) {
  int idx = blockIdx.x * blockDim.x + threadIdx.x;
  if (idx >= BATCH * 64) return;
  zh[idx] = (h16)(eps_t[idx] * expf(0.5f * el[idx]) + em[idx]);
}

// GRU pointwise combine
__global__ void gru_kernel(const float* __restrict__ gi, const float* __restrict__ gh,
                           const float* __restrict__ h, float* __restrict__ hnew) {
  int idx = blockIdx.x * blockDim.x + threadIdx.x;
  if (idx >= BATCH * 256) return;
  int b = idx >> 8, j = idx & 255;
  const float* gib = gi + (size_t)b * 768;
  const float* ghb = gh + (size_t)b * 768;
  float r  = 1.0f / (1.0f + expf(-(gib[j] + ghb[j])));
  float zg = 1.0f / (1.0f + expf(-(gib[256 + j] + ghb[256 + j])));
  float n  = tanhf(gib[512 + j] + r * ghb[512 + j]);
  hnew[idx] = (1.0f - zg) * n + zg * h[idx];
}

// ce -= sum(g_t * g_comb)
__global__ void ce_kernel(const float* __restrict__ gt, const float* __restrict__ gc,
                          float* __restrict__ acc) {
  int idx = blockIdx.x * blockDim.x + threadIdx.x;
  float v = (idx < BATCH * 64) ? gt[idx] * gc[idx] : 0.0f;
  v = wave_sum(v);
  if ((threadIdx.x & 31) == 0) atomicAdd(acc, -v);
}

// kld += mean_b 0.5*(sum(pl-el) - 64 + sum(exp(el-pl)) + sum((em-pm)^2/exp(pl)))
__global__ void kld_kernel(const float* __restrict__ em, const float* __restrict__ el,
                           const float* __restrict__ pm, const float* __restrict__ pl,
                           float* __restrict__ acc) {
  int b = blockIdx.x * blockDim.x + threadIdx.x;
  float c = 0.0f;
  if (b < BATCH) {
    float x1 = 0.0f, x2 = 0.0f, x3 = 0.0f;
    for (int j = 0; j < 64; ++j) {
      int i = b * 64 + j;
      float d = el[i] - pl[i];
      x1 -= d;
      x2 += expf(d);
      float dm = em[i] - pm[i];
      x3 += dm * dm * expf(-pl[i]);
    }
    c = 0.5f * (x1 - 64.0f + x2 + x3) * (1.0f / BATCH);
  }
  c = wave_sum(c);
  if ((threadIdx.x & 31) == 0) atomicAdd(acc, c);
}

// fused dec_mean/dec_logvar (256 -> 2 each) + NLL accumulation
__global__ void nll_kernel(const float* __restrict__ dech,
                           const float* __restrict__ Wm, const float* __restrict__ bm,
                           const float* __restrict__ Wl, const float* __restrict__ bl,
                           const float* __restrict__ xt, float* __restrict__ acc) {
  int b = blockIdx.x * blockDim.x + threadIdx.x;
  float c = 0.0f;
  if (b < BATCH) {
    float dm0 = bm[0], dm1 = bm[1], dl0 = bl[0], dl1 = bl[1];
    const float* row = dech + (size_t)b * 256;
    for (int k = 0; k < 256; ++k) {
      float hv = row[k];
      dm0 += hv * Wm[k * 2];     dm1 += hv * Wm[k * 2 + 1];
      dl0 += hv * Wl[k * 2];     dl1 += hv * Wl[k * 2 + 1];
    }
    float x0 = xt[b * 2], x1 = xt[b * 2 + 1];
    float n1 = (x0 - dm0) * (x0 - dm0) * expf(-dl0)
             + (x1 - dm1) * (x1 - dm1) * expf(-dl1);
    float n3 = dl0 + dl1;
    c = 0.5f * (n1 + 3.6757541328186907f /* 2*log(2*pi) */ + n3) * (1.0f / BATCH);
  }
  c = wave_sum(c);
  if ((threadIdx.x & 31) == 0) atomicAdd(acc, c);
}

__global__ void init_kernel(float* __restrict__ h0, float* __restrict__ scal) {
  int idx = blockIdx.x * blockDim.x + threadIdx.x;
  if (idx < 3) scal[idx] = 0.0f;
  if (idx < BATCH * 256) h0[idx] = 0.0f;
}

__global__ void final_kernel(const float* __restrict__ scal, const float* __restrict__ h,
                             float* __restrict__ out) {
  int idx = blockIdx.x * blockDim.x + threadIdx.x;
  if (idx < 3) out[idx] = scal[idx];
  int n = idx - 3;
  if (n >= 0 && n < BATCH * 256) out[idx] = h[n];
}

// --------------------------------------------------------------------- host

enum {
  IN_TRAJ = 0, IN_TRAJ_REL, IN_GOALS, IN_SSE, IN_ADJ, IN_EPS,
  P_DEC1_W, P_DEC1_B, P_DEC2_W, P_DEC2_B, P_DECG1_W, P_DECG1_B, P_DECG2_W, P_DECG2_B,
  P_DECLV_W, P_DECLV_B, P_DECM_W, P_DECM_B, P_ENC1_W, P_ENC1_B, P_ENC2_W, P_ENC2_B,
  P_ENCLV_W, P_ENCLV_B, P_ENCM_W, P_ENCM_B, P_GG1_W, P_GG1_B, P_GG2_W, P_GG2_B,
  P_GH1_W, P_GH1_B, P_GH2_W, P_GH2_B, P_GRU_WHH, P_GRU_WIH, P_GRU_BHH, P_GRU_BIH,
  P_LGG_W, P_LGG_B, P_LGH_W, P_LGH_B, P_PHIX1_W, P_PHIX1_B, P_PHIX2_W, P_PHIX2_B,
  P_PHIZ1_W, P_PHIZ1_B, P_PHIZ2_W, P_PHIZ2_B, P_PR1_W, P_PR1_B, P_PR2_W, P_PR2_B,
  P_PRLV_W, P_PRLV_B, P_PRM_W, P_PRM_B
};

extern "C" void kernel_launch(void* const* d_in, const int* in_sizes, int n_in,
                              void* d_out, int out_size, void* d_ws, size_t ws_size,
                              hipStream_t stream) {
  (void)in_sizes; (void)n_in; (void)out_size; (void)ws_size;

  uintptr_t base = (uintptr_t)d_ws;
  size_t off = 0;
  auto alloc = [&](size_t bytes) -> void* {
    off = (off + 255) & ~(size_t)255;
    void* p = (void*)(base + off);
    off += bytes;
    return p;
  };
  auto fbuf = [&](int w) { return (float*)alloc((size_t)BATCH * w * sizeof(float)); };
  auto hbuf = [&](int w) { return (h16*)alloc((size_t)BATCH * w * sizeof(h16)); };
  auto fptr = [&](int i) { return (const float*)d_in[i]; };

  auto elem = [&](int total) { return dim3((total + 255) / 256); };

  // -------- per-launch f16 weight conversion (deterministic, graph-safe)
  auto wc = [&](int widx, int K, int N, int Kpad) -> h16* {
    h16* p = (h16*)alloc((size_t)Kpad * N * sizeof(h16));
    wconv_kernel<<<elem(Kpad * N), 256, 0, stream>>>(fptr(widx), p, K, N, Kpad);
    return p;
  };
  h16* w_decg1 = wc(P_DECG1_W, 352, 256, 352);
  h16* w_decg2 = wc(P_DECG2_W, 256,  64, 256);
  h16* w_gg1   = wc(P_GG1_W,    64, 256,  64);
  h16* w_gg2   = wc(P_GG2_W,   256,  64, 256);
  h16* w_lgg   = wc(P_LGG_W,   128,  64, 128);
  h16* w_phix1 = wc(P_PHIX1_W,   2, 256,  32);
  h16* w_phix2 = wc(P_PHIX2_W, 256, 256, 256);
  h16* w_enc1  = wc(P_ENC1_W,  576, 256, 576);
  h16* w_enc2  = wc(P_ENC2_W,  256, 256, 256);
  h16* w_encm  = wc(P_ENCM_W,  256,  64, 256);
  h16* w_enclv = wc(P_ENCLV_W, 256,  64, 256);
  h16* w_pr1   = wc(P_PR1_W,   320, 256, 320);
  h16* w_pr2   = wc(P_PR2_W,   256, 256, 256);
  h16* w_prm   = wc(P_PRM_W,   256,  64, 256);
  h16* w_prlv  = wc(P_PRLV_W,  256,  64, 256);
  h16* w_phiz1 = wc(P_PHIZ1_W,  64, 256,  64);
  h16* w_phiz2 = wc(P_PHIZ2_W, 256, 256, 256);
  h16* w_dec1  = wc(P_DEC1_W,  608, 256, 608);
  h16* w_dec2  = wc(P_DEC2_W,  256, 256, 256);
  h16* w_wih   = wc(P_GRU_WIH, 512, 768, 512);
  h16* w_whh   = wc(P_GRU_WHH, 256, 768, 256);
  h16* w_gh1   = wc(P_GH1_W,   256, 256, 256);
  h16* w_gh2   = wc(P_GH2_W,   256, 256, 256);
  h16* w_lgh   = wc(P_LGH_W,   512, 256, 512);

  // -------- f32 activation buffers
  float* scal  = (float*)alloc(3 * sizeof(float));
  float* h_a   = fbuf(256); float* h_b = fbuf(256);
  float* dt    = fbuf(32);
  float* dgh   = fbuf(256); float* tmp256 = fbuf(256);
  float* t64   = fbuf(64);  float* gdec = fbuf(64);
  float* y256  = fbuf(256); float* g1 = fbuf(256);
  float* y64   = fbuf(64);  float* ggraph = fbuf(64); float* gcomb = fbuf(64);
  float* phix  = fbuf(256);
  float* em    = fbuf(64);  float* el = fbuf(64);
  float* pm    = fbuf(64);  float* pl = fbuf(64);
  float* phiz  = fbuf(256); float* dech = fbuf(256);
  float* gi    = fbuf(768); float* ghh = fbuf(768);
  float* hnew  = fbuf(256); float* hg1 = fbuf(256); float* hgraph = fbuf(256);

  // -------- f16 A-matrix staging buffers
  h16* a352 = hbuf(352); h16* a128 = hbuf(128); h16* a32 = hbuf(32);
  h16* a576 = hbuf(576); h16* a320 = hbuf(320); h16* a608 = hbuf(608);
  h16* a512a = hbuf(512); h16* a512b = hbuf(512);
  h16* a64a = hbuf(64);  h16* a64b = hbuf(64);
  h16* aA = hbuf(256); h16* aB = hbuf(256); h16* aC = hbuf(256); h16* aD = hbuf(256);
  h16* aE = hbuf(256); h16* aF = hbuf(256); h16* aG = hbuf(256); h16* aH = hbuf(256);
  h16* aI = hbuf(256); h16* aJ = hbuf(256); h16* aK = hbuf(256); h16* aL = hbuf(256);

  auto gemm = [&](const h16* A, const h16* W, const float* bias, float* C, h16* Ch,
                  int Kpad, int N, int act) {
    gemm_wmma_kernel<<<dim3(N / 64, BATCH / 16), 32, 0, stream>>>(A, W, bias, C, Ch,
                                                                  Kpad, N, act);
  };
  auto gcn = [&](const float* adjt, const float* X, const float* bias, float* Y, h16* Yh,
                 int N, int relu) {
    gcn_wmma_kernel<<<dim3(N / 64, SBLK), 32, 0, stream>>>(adjt, X, bias, Y, Yh, N, relu);
  };
  auto pack = [&](const float* s0, int w0, const float* s1, int w1,
                  const float* s2, int w2, const float* s3, int w3, h16* A, int Kpad) {
    pack_cat_kernel<<<elem(BATCH * Kpad), 256, 0, stream>>>(s0, w0, s1, w1, s2, w2,
                                                            s3, w3, A, Kpad);
  };

  const float* traj     = fptr(IN_TRAJ);
  const float* traj_rel = fptr(IN_TRAJ_REL);
  const float* goals    = fptr(IN_GOALS);
  const float* adj      = fptr(IN_ADJ);
  const float* eps      = fptr(IN_EPS);

  init_kernel<<<elem(BATCH * 256), 256, 0, stream>>>(h_a, scal);

  float* h = h_a; float* h_next = h_b;
  for (int ti = 0; ti < T_STEPS - 1; ++ti) {
    const int t = ti + 1;
    const float* xt    = traj_rel + (size_t)t * BATCH * 2;
    const float* gt    = goals + (size_t)t * BATCH * 64;
    const float* gprev = goals + (size_t)ti * BATCH * 64;
    const float* adjt  = adj + (size_t)t * BATCH * BATCH;
    const float* epst  = eps + (size_t)ti * BATCH * 64;
    const float* trajt = traj + (size_t)t * BATCH * 2;

    // d_t and decoder-goal branch
    dt_kernel<<<elem(BATCH * 32), 256, 0, stream>>>(trajt, dt);
    pack(dt, 32, h, 256, gprev, 64, nullptr, 0, a352, 352);
    gemm(a352, w_decg1, fptr(P_DECG1_B), dgh, aA, 352, 256, 1);
    gemm(aA, w_decg2, fptr(P_DECG2_B), t64, nullptr, 256, 64, 0);
    softmax64_kernel<<<BATCH, 32, 0, stream>>>(t64, gdec, a64a);
    // goal GCN
    gemm(a64a, w_gg1, nullptr, y256, nullptr, 64, 256, 0);
    gcn(adjt, y256, fptr(P_GG1_B), g1, aB, 256, 1);
    gemm(aB, w_gg2, nullptr, y64, nullptr, 256, 64, 0);
    gcn(adjt, y64, fptr(P_GG2_B), ggraph, nullptr, 64, 0);
    pack(gdec, 64, ggraph, 64, nullptr, 0, nullptr, 0, a128, 128);
    gemm(a128, w_lgg, fptr(P_LGG_B), t64, nullptr, 128, 64, 0);
    softmax64_kernel<<<BATCH, 32, 0, stream>>>(t64, gcomb, nullptr);
    ce_kernel<<<elem(BATCH * 64), 256, 0, stream>>>(gt, gcomb, scal + 2);
    // phi_x
    pack(xt, 2, nullptr, 0, nullptr, 0, nullptr, 0, a32, 32);
    gemm(a32, w_phix1, fptr(P_PHIX1_B), tmp256, aC, 32, 256, 2);
    gemm(aC, w_phix2, fptr(P_PHIX2_B), phix, nullptr, 256, 256, 2);
    // encoder
    pack(phix, 256, gt, 64, h, 256, nullptr, 0, a576, 576);
    gemm(a576, w_enc1, fptr(P_ENC1_B), tmp256, aD, 576, 256, 2);
    gemm(aD, w_enc2, fptr(P_ENC2_B), tmp256, aE, 256, 256, 2);
    gemm(aE, w_encm, fptr(P_ENCM_B), em, nullptr, 256, 64, 0);
    gemm(aE, w_enclv, fptr(P_ENCLV_B), el, nullptr, 256, 64, 0);
    // prior
    pack(gt, 64, h, 256, nullptr, 0, nullptr, 0, a320, 320);
    gemm(a320, w_pr1, fptr(P_PR1_B), tmp256, aF, 320, 256, 2);
    gemm(aF, w_pr2, fptr(P_PR2_B), tmp256, aG, 256, 256, 2);
    gemm(aG, w_prm, fptr(P_PRM_B), pm, nullptr, 256, 64, 0);
    gemm(aG, w_prlv, fptr(P_PRLV_B), pl, nullptr, 256, 64, 0);
    // z, phi_z
    zpack_kernel<<<elem(BATCH * 64), 256, 0, stream>>>(epst, el, em, a64b);
    gemm(a64b, w_phiz1, fptr(P_PHIZ1_B), tmp256, aH, 64, 256, 2);
    gemm(aH, w_phiz2, fptr(P_PHIZ2_B), phiz, nullptr, 256, 256, 2);
    // decoder
    pack(dt, 32, gt, 64, phiz, 256, h, 256, a608, 608);
    gemm(a608, w_dec1, fptr(P_DEC1_B), tmp256, aI, 608, 256, 2);
    gemm(aI, w_dec2, fptr(P_DEC2_B), dech, nullptr, 256, 256, 2);
    nll_kernel<<<elem(BATCH), 256, 0, stream>>>(dech, fptr(P_DECM_W), fptr(P_DECM_B),
                                                fptr(P_DECLV_W), fptr(P_DECLV_B),
                                                xt, scal + 1);
    kld_kernel<<<elem(BATCH), 256, 0, stream>>>(em, el, pm, pl, scal + 0);
    // GRU
    pack(phix, 256, phiz, 256, nullptr, 0, nullptr, 0, a512a, 512);
    gemm(a512a, w_wih, fptr(P_GRU_BIH), gi, nullptr, 512, 768, 0);
    pack(h, 256, nullptr, 0, nullptr, 0, nullptr, 0, aJ, 256);
    gemm(aJ, w_whh, fptr(P_GRU_BHH), ghh, nullptr, 256, 768, 0);
    gru_kernel<<<elem(BATCH * 256), 256, 0, stream>>>(gi, ghh, h, hnew);
    // hidden GCN + combine
    pack(hnew, 256, nullptr, 0, nullptr, 0, nullptr, 0, aK, 256);
    gemm(aK, w_gh1, nullptr, y256, nullptr, 256, 256, 0);
    gcn(adjt, y256, fptr(P_GH1_B), hg1, aL, 256, 1);
    gemm(aL, w_gh2, nullptr, y256, nullptr, 256, 256, 0);
    gcn(adjt, y256, fptr(P_GH2_B), hgraph, nullptr, 256, 0);
    pack(hgraph, 256, hnew, 256, nullptr, 0, nullptr, 0, a512b, 512);
    gemm(a512b, w_lgh, fptr(P_LGH_B), h_next, nullptr, 512, 256, 0);

    float* tmp = h; h = h_next; h_next = tmp;
  }

  final_kernel<<<elem(3 + BATCH * 256), 256, 0, stream>>>(scal, h, (float*)d_out);
}
